// LightChannelAttention_31129922961584
// MI455X (gfx1250) — matrix operand order repaired and checked by reading, not compiled, not run on hardware
//
#include <hip/hip_runtime.h>
#include <hip/hip_bf16.h>

typedef __attribute__((ext_vector_type(16))) _Float16 v16h;
typedef __attribute__((ext_vector_type(8)))  _Float16 v8h;
typedef __attribute__((ext_vector_type(8)))  float    v8f;

#define LDAS 40   // 64x32 A tile row stride (halfs), padded
#define BCH  24   // B fragment chunk stride (halfs): 16 data + 8 pad, 48B (16B-aligned)

// ---------------- CDNA5 async memory->LDS helpers (ASYNCcnt) ----------------

// One 16B async copy global->LDS (no VGPR data path).
__device__ __forceinline__ void async_b128(unsigned lds_off, const _Float16* src) {
  unsigned long long ga = (unsigned long long)(uintptr_t)src;
  asm volatile("global_load_async_to_lds_b128 %0, %1, off"
               :: "v"(lds_off), "v"(ga) : "memory");
}

// Two 16B async copies; INST_OFFSET applies to both LDS and global addresses.
__device__ __forceinline__ void async_b128_x2(unsigned lds_off, const _Float16* src) {
  unsigned long long ga = (unsigned long long)(uintptr_t)src;
  asm volatile("global_load_async_to_lds_b128 %0, %1, off\n\t"
               "global_load_async_to_lds_b128 %0, %1, off offset:16"
               :: "v"(lds_off), "v"(ga) : "memory");
}

__device__ __forceinline__ void wait_async0() {
  asm volatile("s_wait_asynccnt 0x0" ::: "memory");
}

// ---------------- WMMA fragment helpers (wave32, 16x16x32 f16) ----------------

// A tile: 16 rows (m) x 32 cols (k), row-major, stride LDAS. Two b128 loads per lane.
__device__ __forceinline__ v16h frag_a(const _Float16* As) {
  int lane = threadIdx.x & 31;
  int m  = lane & 15;
  int kh = (lane >> 4) << 4;          // lanes 0-15: K 0-15; lanes 16-31: K 16-31
  const _Float16* row = As + m * LDAS + kh;
  v8h lo = *(const v8h*)(row);        // K kh+0..7
  v8h hi = *(const v8h*)(row + 8);    // K kh+8..15
  return __builtin_shufflevector(lo, hi, 0,1,2,3,4,5,6,7,8,9,10,11,12,13,14,15);
}

// B operand in fragment order: chunk (s*32+lane) holds 16 consecutive K for one N.
__device__ __forceinline__ v16h frag_bx(const _Float16* Bx, int s) {
  int lane = threadIdx.x & 31;
  const _Float16* c = Bx + (s * 32 + lane) * BCH;
  v8h lo = *(const v8h*)c;
  v8h hi = *(const v8h*)(c + 8);
  return __builtin_shufflevector(lo, hi, 0,1,2,3,4,5,6,7,8,9,10,11,12,13,14,15);
}

__device__ __forceinline__ void mma_step(const _Float16* As, const _Float16* Bx,
                                         v8f acc[4], int wv) {
  int mi = wv & 3;
  int s0 = (wv >> 2) << 2;            // n-subtile base (0 or 4)
  v16h a = frag_a(As + mi * 16 * LDAS);
#pragma unroll
  for (int j = 0; j < 4; ++j) {
    v16h b = frag_bx(Bx, s0 + j);
    acc[j] = __builtin_amdgcn_wmma_f32_16x16x32_f16(false, a, false, b,
                                                    (short)0, acc[j], false, false);
  }
}

// Cooperative async fill of the fragment-ordered B tile from a fragment-packed
// global buffer: pkB[((k>>5)*N + n)*32 + ((k>>4)&1)*16 + (k&15)].
// Chunk index == threadIdx.x by construction; 32B per thread, no VGPR data path.
__device__ __forceinline__ void fill_bx(_Float16* Bx, const _Float16* pkB,
                                        int kblk, int n0c, int N) {
  int t = threadIdx.x;
  int j = t >> 5, l = t & 31;
  int n = n0c + (j << 4) + (l & 15);
  int khalf = l >> 4;
  const _Float16* src = pkB + (((size_t)kblk * N + n) * 2 + khalf) * 16;
  async_b128_x2((unsigned)(uintptr_t)(Bx + t * BCH), src);
}

// Row-major A tile fill: one 16B async copy per thread.
__device__ __forceinline__ void fill_a_rm(_Float16* As, const _Float16* A,
                                          size_t lda, int m0, int kblk) {
  int t = threadIdx.x;
  int m = t >> 2, kk0 = (t & 3) << 3;
  async_b128((unsigned)(uintptr_t)&As[m * LDAS + kk0],
             &A[(size_t)(m0 + m) * lda + ((size_t)kblk << 5) + kk0]);
}

// im2col A tile fill with k = r*C + c ordering (filter tap major, channel minor).
// A 64-row M-tile is exactly one image row -> h block-constant, w == m.
// Per chunk: (dh,dw) uniform; bounds check uniform per thread; 8 channel-strided
// u16 loads (lane-coalesced) + one b128 LDS store.
template <int CBL2>   // log2(C/32): 3 for C=256, 2 for C=128
__device__ __forceinline__ void fill_a_conv(_Float16* As, const _Float16* xb,
                                            int kblk, int h) {
  int t = threadIdx.x;
  int r  = kblk >> CBL2;
  int c0 = (kblk & ((1 << CBL2) - 1)) << 5;
  int dh = r / 3;
  int dw = r - dh * 3 - 1;
  dh -= 1;
  int m = t >> 2, kk0 = (t & 3) << 3;
  int hh = h + dh, ww = m + dw;
  v8h val = {};
  if ((unsigned)hh < 64u && (unsigned)ww < 64u) {
    const _Float16* src = xb + ((size_t)(c0 + kk0) << 12) + hh * 64 + ww;
#pragma unroll
    for (int i = 0; i < 8; ++i) val[i] = src[(size_t)i << 12];
  }
  *(v8h*)&As[m * LDAS + kk0] = val;
}

__device__ __forceinline__ size_t pk_index(int k, int n, int N) {
  return (((size_t)(k >> 5) * N + n) * 2 + ((k >> 4) & 1)) * 16 + (k & 15);
}

// ---------------- prep: f32->f16 + fragment-packed copies of x ----------------

__global__ __launch_bounds__(256) void convert_f16_kernel(const float* __restrict__ src,
    _Float16* __restrict__ xh, _Float16* __restrict__ pkxg, _Float16* __restrict__ pkxc) {
  size_t base = ((size_t)blockIdx.x * 256 + threadIdx.x) * 4;
  int pos0 = (int)(base & 4095);
  int c    = (int)((base >> 12) & 255);
  int b    = (int)(base >> 20);
  size_t pb = (size_t)b * 256 * 4096;
#pragma unroll
  for (int i = 0; i < 4; ++i) {
    _Float16 v = (_Float16)src[base + i];
    int pos = pos0 + i;
    xh[base + i] = v;
    pkxg[pb + pk_index(pos, c, 256)]  = v;   // K = pos (4096), N = d (256)
    pkxc[pb + pk_index(c, pos, 4096)] = v;   // K = d (256),  N = pos (4096)
  }
}

// w_reduce [128,256,3,3] -> fragment-packed w1p, k = r*256 + c; BN1 folded; b1f[128]
__global__ __launch_bounds__(256) void pack_w1_kernel(const float* __restrict__ w,
    const float* __restrict__ sc, const float* __restrict__ bi,
    const float* __restrict__ mu, const float* __restrict__ va,
    _Float16* __restrict__ w1p, float* __restrict__ b1f) {
  int g = blockIdx.x * 256 + threadIdx.x;
  if (g >= 2304 * 128) return;
  int o = g & 127;
  int k = g >> 7;
  int r = k >> 8, c = k & 255;
  float inv = sc[o] * rsqrtf(va[o] + 1e-5f);
  w1p[pk_index(k, o, 128)] = (_Float16)(w[((size_t)o * 256 + c) * 9 + r] * inv);
  if (g < 128) b1f[g] = bi[g] - mu[g] * inv;
}

// w_fusion [256,128,3,3] -> fragment-packed w2p, k = r*128 + c; BN2 folded; b2f[256]
__global__ __launch_bounds__(256) void pack_w2_kernel(const float* __restrict__ w,
    const float* __restrict__ sc, const float* __restrict__ bi,
    const float* __restrict__ mu, const float* __restrict__ va,
    _Float16* __restrict__ w2p, float* __restrict__ b2f) {
  int g = blockIdx.x * 256 + threadIdx.x;
  if (g >= 1152 * 256) return;
  int o = g & 255;
  int k = g >> 8;
  int r = k >> 7, c = k & 127;
  float inv = sc[o] * rsqrtf(va[o] + 1e-5f);
  w2p[pk_index(k, o, 256)] = (_Float16)(w[((size_t)o * 128 + c) * 9 + r] * inv);
  if (g < 256) b2f[g] = bi[g] - mu[g] * inv;
}

// ---------------- conv1: implicit GEMM, M=4096(spatial) N=128 K=2304 ----------------

__global__ __launch_bounds__(256) void conv1_kernel(const _Float16* __restrict__ xh,
    const _Float16* __restrict__ w1p, const float* __restrict__ b1f,
    _Float16* __restrict__ reduced) {
  __shared__ __align__(16) _Float16 As[2][64 * LDAS];
  __shared__ __align__(16) _Float16 Bx[2][256 * BCH];
  const int b  = blockIdx.z;
  const int m0 = blockIdx.x * 64;
  const int h  = m0 >> 6;
  const int t  = threadIdx.x;
  const int wv = t >> 5;
  const _Float16* xb = xh + (size_t)b * 256 * 4096;
  v8f acc[4] = {};
  const int NBLK = 72;  // 2304/32
  fill_a_conv<3>(As[0], xb, 0, h);
  fill_bx(Bx[0], w1p, 0, 0, 128);
  wait_async0();
  __syncthreads();
  for (int i = 0; i < NBLK; ++i) {
    int cur = i & 1;
    if (i + 1 < NBLK) {
      fill_a_conv<3>(As[cur ^ 1], xb, i + 1, h);
      fill_bx(Bx[cur ^ 1], w1p, i + 1, 0, 128);
    }
    mma_step(As[cur], Bx[cur], acc, wv);
    wait_async0();
    __syncthreads();
  }
  int lane = t & 31, mi = wv & 3, nb = (wv >> 2) << 6;
  int cn = lane & 15, rh = (lane >> 4) << 3;
#pragma unroll
  for (int j = 0; j < 4; ++j) {
    int n = nb + j * 16 + cn;
    float bias = b1f[n];
    v8h st;
#pragma unroll
    for (int v = 0; v < 8; ++v) {
      float val = acc[j][v] + bias;
      st[v] = (_Float16)(val > 0.f ? val : 0.f);
    }
    *(v8h*)&reduced[((size_t)b * 128 + n) * 4096 + m0 + mi * 16 + rh] = st;
  }
}

// ---------------- 2x2 avg pool -> p row-major + two fragment-packed copies ----------------

__global__ __launch_bounds__(256) void pool_kernel(const _Float16* __restrict__ reduced,
    _Float16* __restrict__ p, _Float16* __restrict__ pks, _Float16* __restrict__ pkr) {
  size_t g = (size_t)blockIdx.x * 256 + threadIdx.x;   // ((b*1024+np)*128 + c)
  int c  = g & 127;
  int np = (g >> 7) & 1023;
  int b  = g >> 17;
  int h0 = (np >> 5) << 1, w0 = (np & 31) << 1;
  const _Float16* rb = reduced + ((size_t)b * 128 + c) * 4096;
  float s = (float)rb[h0 * 64 + w0] + (float)rb[h0 * 64 + w0 + 1] +
            (float)rb[(h0 + 1) * 64 + w0] + (float)rb[(h0 + 1) * 64 + w0 + 1];
  _Float16 v = (_Float16)(0.25f * s);
  size_t pb = (size_t)b * 1024 * 128;
  p[g] = v;
  pks[pb + pk_index(c, np, 1024)] = v;   // B of S-GEMM: K=c(128),  N=m(1024)
  pkr[pb + pk_index(np, c, 128)]  = v;   // B of R-GEMM: K=m(1024), N=c(128)
}

// ---------------- S = p @ p^T : M=N=1024 K=128, f16 out ----------------

__global__ __launch_bounds__(256) void gemm_s_kernel(const _Float16* __restrict__ p,
    const _Float16* __restrict__ pks, _Float16* __restrict__ S) {
  __shared__ __align__(16) _Float16 As[2][64 * LDAS];
  __shared__ __align__(16) _Float16 Bx[2][256 * BCH];
  const int b = blockIdx.z, m0 = blockIdx.x * 64, n0c = blockIdx.y * 128;
  const int t = threadIdx.x, wv = t >> 5;
  const _Float16* pa = p   + (size_t)b * 1024 * 128;
  const _Float16* pb = pks + (size_t)b * 1024 * 128;
  v8f acc[4] = {};
  const int NBLK = 4;
  fill_a_rm(As[0], pa, 128, m0, 0);
  fill_bx(Bx[0], pb, 0, n0c, 1024);
  wait_async0();
  __syncthreads();
  for (int i = 0; i < NBLK; ++i) {
    int cur = i & 1;
    if (i + 1 < NBLK) {
      fill_a_rm(As[cur ^ 1], pa, 128, m0, i + 1);
      fill_bx(Bx[cur ^ 1], pb, i + 1, n0c, 1024);
    }
    mma_step(As[cur], Bx[cur], acc, wv);
    wait_async0();
    __syncthreads();
  }
  int lane = t & 31, mi = wv & 3, nb = (wv >> 2) << 6;
  int cn = lane & 15, rh = (lane >> 4) << 3;
#pragma unroll
  for (int j = 0; j < 4; ++j)
#pragma unroll
    for (int v = 0; v < 8; ++v) {
      int mg = m0 + mi * 16 + rh + v, ng = n0c + nb + j * 16 + cn;
      S[((size_t)b * 1024 + mg) * 1024 + ng] = (_Float16)acc[j][v];
    }
}

// ---------------- in-place row softmax, width 1024 ----------------

__global__ __launch_bounds__(256) void softmax1024_kernel(_Float16* __restrict__ S) {
  __shared__ float sm[256];
  size_t base = (size_t)blockIdx.x * 1024;
  int t = threadIdx.x;
  float v[4], mx = -1e30f;
#pragma unroll
  for (int j = 0; j < 4; ++j) { v[j] = (float)S[base + t * 4 + j]; mx = fmaxf(mx, v[j]); }
  sm[t] = mx; __syncthreads();
  for (int s = 128; s > 0; s >>= 1) { if (t < s) sm[t] = fmaxf(sm[t], sm[t + s]); __syncthreads(); }
  mx = sm[0]; __syncthreads();
  float sum = 0.f;
#pragma unroll
  for (int j = 0; j < 4; ++j) { v[j] = __expf(v[j] - mx); sum += v[j]; }
  sm[t] = sum; __syncthreads();
  for (int s = 128; s > 0; s >>= 1) { if (t < s) sm[t] += sm[t + s]; __syncthreads(); }
  float inv = 1.f / sm[0];
#pragma unroll
  for (int j = 0; j < 4; ++j) S[base + t * 4 + j] = (_Float16)(v[j] * inv);
}

// ---------------- refined_small = att @ p : M=1024 N=128 K=1024 -> rs[b,128,32,32] ----------------

__global__ __launch_bounds__(256) void gemm_r_kernel(const _Float16* __restrict__ S,
    const _Float16* __restrict__ pkr, _Float16* __restrict__ rs) {
  __shared__ __align__(16) _Float16 As[2][64 * LDAS];
  __shared__ __align__(16) _Float16 Bx[2][256 * BCH];
  const int b = blockIdx.z, m0 = blockIdx.x * 64;
  const int t = threadIdx.x, wv = t >> 5;
  const _Float16* Sb = S   + (size_t)b * 1024 * 1024;
  const _Float16* pb = pkr + (size_t)b * 1024 * 128;
  v8f acc[4] = {};
  const int NBLK = 32;
  fill_a_rm(As[0], Sb, 1024, m0, 0);
  fill_bx(Bx[0], pb, 0, 0, 128);
  wait_async0();
  __syncthreads();
  for (int i = 0; i < NBLK; ++i) {
    int cur = i & 1;
    if (i + 1 < NBLK) {
      fill_a_rm(As[cur ^ 1], Sb, 1024, m0, i + 1);
      fill_bx(Bx[cur ^ 1], pb, i + 1, 0, 128);
    }
    mma_step(As[cur], Bx[cur], acc, wv);
    wait_async0();
    __syncthreads();
  }
  int lane = t & 31, mi = wv & 3, nb = (wv >> 2) << 6;
  int cn = lane & 15, rh = (lane >> 4) << 3;
#pragma unroll
  for (int j = 0; j < 4; ++j) {
    int c = nb + j * 16 + cn;           // channel 0..127
    v8h st;
#pragma unroll
    for (int v = 0; v < 8; ++v) st[v] = (_Float16)acc[j][v];
    *(v8h*)&rs[((size_t)b * 128 + c) * 1024 + m0 + mi * 16 + rh] = st;
  }
}

// ---------------- bilinear upsample 32->64 (half-pixel, edge clamp) ----------------

__global__ __launch_bounds__(256) void upsample_kernel(const _Float16* __restrict__ rs,
                                                       _Float16* __restrict__ ups) {
  size_t g = (size_t)blockIdx.x * 256 + threadIdx.x;   // ((b*128+c)*64+h)*64+w
  int w = g & 63, h = (g >> 6) & 63, c = (g >> 12) & 127, b = g >> 19;
  float fy = h * 0.5f - 0.25f, fx = w * 0.5f - 0.25f;
  fy = fminf(fmaxf(fy, 0.f), 31.f);
  fx = fminf(fmaxf(fx, 0.f), 31.f);
  int y0 = (int)fy, x0 = (int)fx;
  int y1 = min(y0 + 1, 31), x1 = min(x0 + 1, 31);
  float wy = fy - y0, wx = fx - x0;
  const _Float16* rb = rs + ((size_t)b * 128 + c) * 1024;
  float v00 = (float)rb[y0 * 32 + x0], v01 = (float)rb[y0 * 32 + x1];
  float v10 = (float)rb[y1 * 32 + x0], v11 = (float)rb[y1 * 32 + x1];
  float val = (1.f - wy) * ((1.f - wx) * v00 + wx * v01) + wy * ((1.f - wx) * v10 + wx * v11);
  ups[g] = (_Float16)val;
}

// ---------------- G = a @ a^T : M=N=256 K=4096, f16 out ----------------

__global__ __launch_bounds__(256) void gemm_g_kernel(const _Float16* __restrict__ xh,
    const _Float16* __restrict__ pkxg, _Float16* __restrict__ G) {
  __shared__ __align__(16) _Float16 As[2][64 * LDAS];
  __shared__ __align__(16) _Float16 Bx[2][256 * BCH];
  const int b = blockIdx.z, m0 = blockIdx.x * 64, n0c = blockIdx.y * 128;
  const int t = threadIdx.x, wv = t >> 5;
  const _Float16* xa = xh   + (size_t)b * 256 * 4096;
  const _Float16* xg = pkxg + (size_t)b * 256 * 4096;
  v8f acc[4] = {};
  const int NBLK = 128;
  fill_a_rm(As[0], xa, 4096, m0, 0);
  fill_bx(Bx[0], xg, 0, n0c, 256);
  wait_async0();
  __syncthreads();
  for (int i = 0; i < NBLK; ++i) {
    int cur = i & 1;
    if (i + 1 < NBLK) {
      fill_a_rm(As[cur ^ 1], xa, 4096, m0, i + 1);
      fill_bx(Bx[cur ^ 1], xg, i + 1, n0c, 256);
    }
    mma_step(As[cur], Bx[cur], acc, wv);
    wait_async0();
    __syncthreads();
  }
  int lane = t & 31, mi = wv & 3, nb = (wv >> 2) << 6;
  int cn = lane & 15, rh = (lane >> 4) << 3;
#pragma unroll
  for (int j = 0; j < 4; ++j)
#pragma unroll
    for (int v = 0; v < 8; ++v) {
      int mg = m0 + mi * 16 + rh + v, ng = n0c + nb + j * 16 + cn;
      G[((size_t)b * 256 + mg) * 256 + ng] = (_Float16)acc[j][v];
    }
}

// ---------------- in-place row softmax, width 256 ----------------

__global__ __launch_bounds__(256) void softmax256_kernel(_Float16* __restrict__ G) {
  __shared__ float sm[256];
  size_t base = (size_t)blockIdx.x * 256;
  int t = threadIdx.x;
  float v = (float)G[base + t];
  sm[t] = v; __syncthreads();
  for (int s = 128; s > 0; s >>= 1) { if (t < s) sm[t] = fmaxf(sm[t], sm[t + s]); __syncthreads(); }
  float mx = sm[0]; __syncthreads();
  v = __expf(v - mx);
  sm[t] = v; __syncthreads();
  for (int s = 128; s > 0; s >>= 1) { if (t < s) sm[t] += sm[t + s]; __syncthreads(); }
  G[base + t] = (_Float16)(v / sm[0]);
}

// ---------------- out = x + beta * (attG @ a) : M=256 N=4096 K=256 ----------------

__global__ __launch_bounds__(256) void gemm_ctx_kernel(const _Float16* __restrict__ G,
    const _Float16* __restrict__ pkxc, const float* __restrict__ x,
    const float* __restrict__ beta, float* __restrict__ out) {
  __shared__ __align__(16) _Float16 As[2][64 * LDAS];
  __shared__ __align__(16) _Float16 Bx[2][256 * BCH];
  const int b = blockIdx.z, m0 = blockIdx.x * 64, n0c = blockIdx.y * 128;
  const int t = threadIdx.x, wv = t >> 5;
  const _Float16* Gb = G    + (size_t)b * 256 * 256;
  const _Float16* xc = pkxc + (size_t)b * 256 * 4096;
  v8f acc[4] = {};
  const int NBLK = 8;
  fill_a_rm(As[0], Gb, 256, m0, 0);
  fill_bx(Bx[0], xc, 0, n0c, 4096);
  wait_async0();
  __syncthreads();
  for (int i = 0; i < NBLK; ++i) {
    int cur = i & 1;
    if (i + 1 < NBLK) {
      fill_a_rm(As[cur ^ 1], Gb, 256, m0, i + 1);
      fill_bx(Bx[cur ^ 1], xc, i + 1, n0c, 4096);
    }
    mma_step(As[cur], Bx[cur], acc, wv);
    wait_async0();
    __syncthreads();
  }
  float be = beta[0];
  int lane = t & 31, mi = wv & 3, nb = (wv >> 2) << 6;
  int cn = lane & 15, rh = (lane >> 4) << 3;
#pragma unroll
  for (int j = 0; j < 4; ++j)
#pragma unroll
    for (int v = 0; v < 8; ++v) {
      int cg = m0 + mi * 16 + rh + v, ng = n0c + nb + j * 16 + cn;
      size_t idx = ((size_t)b * 256 + cg) * 4096 + ng;
      out[idx] = x[idx] + be * acc[j][v];
    }
}

// ---------------- conv2: implicit GEMM M=4096 N=256 K=1152, accumulates into out ----------------

__global__ __launch_bounds__(256) void conv2_kernel(const _Float16* __restrict__ ups,
    const _Float16* __restrict__ w2p, const float* __restrict__ b2f,
    float* __restrict__ out) {
  __shared__ __align__(16) _Float16 As[2][64 * LDAS];
  __shared__ __align__(16) _Float16 Bx[2][256 * BCH];
  const int b = blockIdx.z, m0 = blockIdx.x * 64, n0c = blockIdx.y * 128;
  const int h = m0 >> 6;
  const int t = threadIdx.x, wv = t >> 5;
  const _Float16* ub = ups + (size_t)b * 128 * 4096;
  v8f acc[4] = {};
  const int NBLK = 36;  // 1152/32
  fill_a_conv<2>(As[0], ub, 0, h);
  fill_bx(Bx[0], w2p, 0, n0c, 256);
  wait_async0();
  __syncthreads();
  for (int i = 0; i < NBLK; ++i) {
    int cur = i & 1;
    if (i + 1 < NBLK) {
      fill_a_conv<2>(As[cur ^ 1], ub, i + 1, h);
      fill_bx(Bx[cur ^ 1], w2p, i + 1, n0c, 256);
    }
    mma_step(As[cur], Bx[cur], acc, wv);
    wait_async0();
    __syncthreads();
  }
  int lane = t & 31, mi = wv & 3, nb = (wv >> 2) << 6;
  int cn = lane & 15, rh = (lane >> 4) << 3;
#pragma unroll
  for (int j = 0; j < 4; ++j) {
    int n = n0c + nb + j * 16 + cn;     // cout 0..255
    float bias = b2f[n];
#pragma unroll
    for (int v = 0; v < 8; ++v) {
      int mg = m0 + mi * 16 + rh + v;
      float val = acc[j][v] + bias;
      val = val > 0.f ? val : 0.f;
      size_t idx = ((size_t)b * 256 + n) * 4096 + mg;
      out[idx] += val;                  // out already holds x + beta*ctx
    }
  }
}

// ---------------- host launcher ----------------

extern "C" void kernel_launch(void* const* d_in, const int* in_sizes, int n_in,
                              void* d_out, int out_size, void* d_ws, size_t ws_size,
                              hipStream_t stream) {
  (void)in_sizes; (void)n_in; (void)out_size; (void)ws_size;
  const float* x    = (const float*)d_in[0];
  const float* wr   = (const float*)d_in[1];
  const float* s1   = (const float*)d_in[2];
  const float* bi1  = (const float*)d_in[3];
  const float* mu1  = (const float*)d_in[4];
  const float* va1  = (const float*)d_in[5];
  const float* wf   = (const float*)d_in[6];
  const float* s2   = (const float*)d_in[7];
  const float* bi2  = (const float*)d_in[8];
  const float* mu2  = (const float*)d_in[9];
  const float* va2  = (const float*)d_in[10];
  const float* beta = (const float*)d_in[11];
  float* out = (float*)d_out;

  char* w = (char*)d_ws;
  size_t off = 0;
  _Float16* xh   = (_Float16*)(w + off); off += (size_t)16 * 256 * 4096 * 2;   // 32 MB
  _Float16* pkxg = (_Float16*)(w + off); off += (size_t)16 * 256 * 4096 * 2;   // 32 MB
  _Float16* pkxc = (_Float16*)(w + off); off += (size_t)16 * 256 * 4096 * 2;   // 32 MB
  _Float16* w1p  = (_Float16*)(w + off); off += (size_t)2304 * 128 * 2;
  float*    b1f  = (float*)   (w + off); off += 128 * 4 + 256;
  _Float16* w2p  = (_Float16*)(w + off); off += (size_t)1152 * 256 * 2;
  float*    b2f  = (float*)   (w + off); off += 256 * 4;
  _Float16* reduced = (_Float16*)(w + off); off += (size_t)16 * 128 * 4096 * 2; // 16 MB
  _Float16* p    = (_Float16*)(w + off); off += (size_t)16 * 1024 * 128 * 2;    // 4 MB
  _Float16* pks  = (_Float16*)(w + off); off += (size_t)16 * 1024 * 128 * 2;    // 4 MB
  _Float16* pkr  = (_Float16*)(w + off); off += (size_t)16 * 1024 * 128 * 2;    // 4 MB
  _Float16* S    = (_Float16*)(w + off); off += (size_t)16 * 1024 * 1024 * 2;   // 32 MB
  _Float16* rs   = (_Float16*)(w + off); off += (size_t)16 * 128 * 1024 * 2;    // 4 MB
  _Float16* ups  = (_Float16*)(w + off); off += (size_t)16 * 128 * 4096 * 2;    // 16 MB
  _Float16* G    = (_Float16*)(w + off); off += (size_t)16 * 256 * 256 * 2;     // 2 MB

  convert_f16_kernel<<<16384, 256, 0, stream>>>(x, xh, pkxg, pkxc);
  pack_w1_kernel<<<1152, 256, 0, stream>>>(wr, s1, bi1, mu1, va1, w1p, b1f);
  pack_w2_kernel<<<1152, 256, 0, stream>>>(wf, s2, bi2, mu2, va2, w2p, b2f);

  conv1_kernel<<<dim3(64, 1, 16), 256, 0, stream>>>(xh, w1p, b1f, reduced);
  pool_kernel<<<8192, 256, 0, stream>>>(reduced, p, pks, pkr);

  gemm_s_kernel<<<dim3(16, 8, 16), 256, 0, stream>>>(p, pks, S);
  softmax1024_kernel<<<16384, 256, 0, stream>>>(S);
  gemm_r_kernel<<<dim3(16, 1, 16), 256, 0, stream>>>(S, pkr, rs);
  upsample_kernel<<<32768, 256, 0, stream>>>(rs, ups);

  gemm_g_kernel<<<dim3(4, 2, 16), 256, 0, stream>>>(xh, pkxg, G);
  softmax256_kernel<<<4096, 256, 0, stream>>>(G);
  gemm_ctx_kernel<<<dim3(4, 32, 16), 256, 0, stream>>>(G, pkxc, x, beta, out);

  conv2_kernel<<<dim3(64, 2, 16), 256, 0, stream>>>(ups, w2p, b2f, out);
}
